// GINE_61864708931975
// MI455X (gfx1250) — matrix operand order repaired and checked by reading, not compiled, not run on hardware
//
#include <hip/hip_runtime.h>

// GINE GNN for MI455X (gfx1250, wave32, WMMA)
// - edge message/scatter: persistent waves, coalesced gather + global_atomic_add_f32 (L2-resident)
// - 64-wide GEMMs: v_wmma_f32_16x16x32_f16, f32 accumulate, B staged transposed in LDS (b128 DS reads)
// - head: Wh pre-transposed/f16-packed to [512][64] in ws -> b128 B loads, both head GEMMs fused
// - BN affine + leaky fused into WMMA A-operand load / epilogue

typedef __attribute__((ext_vector_type(16))) _Float16 v16h;
typedef __attribute__((ext_vector_type(8)))  _Float16 v8h;
typedef __attribute__((ext_vector_type(8)))  float    v8f;

union V16 { v16h v; v8h h[2]; };

static constexpr float NEG_SLOPE = 0.01f;
static constexpr float BN_EPS    = 1e-5f;

__device__ __forceinline__ float leakyf(float v) { return v >= 0.f ? v : NEG_SLOPE * v; }

// load 8 consecutive floats (base 16B-aligned) as two b128s
__device__ __forceinline__ void load8f(const float* __restrict__ b, float* d) {
    float4 x = ((const float4*)b)[0];
    float4 y = ((const float4*)b)[1];
    d[0] = x.x; d[1] = x.y; d[2] = x.z; d[3] = x.w;
    d[4] = y.x; d[5] = y.y; d[6] = y.z; d[7] = y.w;
}

// ---------------------------------------------------------------- utilities
__global__ void k_zero(float* __restrict__ p, int n) {
    for (int i = blockIdx.x * blockDim.x + threadIdx.x; i < n; i += gridDim.x * blockDim.x)
        p[i] = 0.f;
}

// ------------------------------------------------- layer 1 edge path (C=2)
__global__ void k_edge2(const long long* __restrict__ ei, const float* __restrict__ ea,
                        const float* __restrict__ x, const float* __restrict__ eW,
                        const float* __restrict__ eb, float* __restrict__ agg2, int E) {
    int e = blockIdx.x * blockDim.x + threadIdx.x;
    if (e >= E) return;
    long long s = ei[e];
    long long d = ei[(size_t)E + e];
    const float* a = ea + (size_t)e * 7;
    float m0 = eb[0], m1 = eb[1];
#pragma unroll
    for (int k = 0; k < 7; ++k) {
        float av = a[k];
        m0 += av * eW[k * 2 + 0];
        m1 += av * eW[k * 2 + 1];
    }
    m0 += x[s * 2 + 0];
    m1 += x[s * 2 + 1];
    m0 = m0 > 0.f ? m0 : 0.f;
    m1 = m1 > 0.f ? m1 : 0.f;
    atomicAdd(&agg2[d * 2 + 0], m0);
    atomicAdd(&agg2[d * 2 + 1], m1);
}

// h = (x + agg2) @ W1a + b1a  ([N,2] x [2,64]); K=2, scalar FMAs
__global__ void k_gemm1(const float* __restrict__ x, const float* __restrict__ agg2,
                        const float* __restrict__ W, const float* __restrict__ b,
                        float* __restrict__ h, int N) {
    int t = blockIdx.x * blockDim.x + threadIdx.x;
    int n = t >> 6, j = t & 63;
    if (n >= N) return;
    float a0 = x[n * 2 + 0] + agg2[n * 2 + 0];
    float a1 = x[n * 2 + 1] + agg2[n * 2 + 1];
    h[(size_t)n * 64 + j] = a0 * W[j] + a1 * W[64 + j] + b[j];
}

// ------------------------------------------------- edge path (C=64): persistent waves
// Each lane owns channels (lane, lane+32); edge-MLP weights live in registers across edges.
__global__ void k_edge64(const long long* __restrict__ ei, const float* __restrict__ ea,
                         const float* __restrict__ x, const float* __restrict__ eW,
                         const float* __restrict__ eb, float* __restrict__ agg, int E) {
    int lane = threadIdx.x & 31;
    int wid = blockIdx.x * (blockDim.x >> 5) + (threadIdx.x >> 5);
    int nw = gridDim.x * (blockDim.x >> 5);
    int c0 = lane, c1 = lane + 32;
    float w0[7], w1[7];
#pragma unroll
    for (int k = 0; k < 7; ++k) {
        w0[k] = eW[k * 64 + c0];
        w1[k] = eW[k * 64 + c1];
    }
    float b0 = eb[c0], b1 = eb[c1];
    for (int e = wid; e < E; e += nw) {
        long long s = ei[e];
        long long d = ei[(size_t)E + e];
        const float* a = ea + (size_t)e * 7;
        float v0 = b0, v1 = b1;
#pragma unroll
        for (int k = 0; k < 7; ++k) {
            float av = a[k];
            v0 += av * w0[k];
            v1 += av * w1[k];
        }
        v0 += x[s * 64 + c0];
        v1 += x[s * 64 + c1];
        v0 = v0 > 0.f ? v0 : 0.f;
        v1 = v1 > 0.f ? v1 : 0.f;
        atomicAdd(&agg[d * 64 + c0], v0);
        atomicAdd(&agg[d * 64 + c1], v1);
    }
}

// ------------------------------------------------- 64x64 WMMA GEMM (one wave / 16 rows)
// MODE 0: A = p0 + p1 (x + aggr), epilogue = +bias                 (produces pre-BN h)
// MODE 1: A = leaky(p0*scale+shift), epilogue = leaky(+bias)       (BN fused in A-load)
// Weights staged in LDS transposed (WT[n][k], stride 80 halves): B fragment = 2x ds_load_b128.
template <int MODE>
__global__ void k_gemm64(const float* __restrict__ p0, const float* __restrict__ p1,
                         const float* __restrict__ p2, const float* __restrict__ W,
                         const float* __restrict__ bias, float* __restrict__ out, int N) {
    constexpr int WS = 80;  // padded stride in halves: 160B rows, 16B aligned, 2-way banks
    __shared__ _Float16 WT[64 * WS];
    for (int t = threadIdx.x; t < 64 * 64; t += blockDim.x) {
        int k = t >> 6, n = t & 63;
        WT[n * WS + k] = (_Float16)W[t];
    }
    __syncthreads();

    int ntiles = (N + 15) >> 4;
    int tile = blockIdx.x * (blockDim.x >> 5) + (threadIdx.x >> 5);
    if (tile >= ntiles) return;  // whole-wave exit: EXEC all-ones for WMMA

    int lane = threadIdx.x & 31;
    int m = lane & 15;
    int hi = lane >> 4;  // 0/1
    int row = tile * 16 + m;
    if (row >= N) row = N - 1;
    int kh = hi * 8;

    // A per ISA 16-bit 16x32 layout = four contiguous 8-float runs per lane -> b128 loads
    float ar[32];
    const float* r0 = p0 + (size_t)row * 64;
    load8f(r0 + kh, ar);
    load8f(r0 + kh + 16, ar + 8);
    load8f(r0 + kh + 32, ar + 16);
    load8f(r0 + kh + 48, ar + 24);
    if constexpr (MODE == 0) {
        float gg[32];
        const float* r1 = p1 + (size_t)row * 64;
        load8f(r1 + kh, gg);
        load8f(r1 + kh + 16, gg + 8);
        load8f(r1 + kh + 32, gg + 16);
        load8f(r1 + kh + 48, gg + 24);
#pragma unroll
        for (int i = 0; i < 32; ++i) ar[i] += gg[i];
    } else {
        float sc[32], sf[32];
        load8f(p1 + kh, sc);
        load8f(p1 + kh + 16, sc + 8);
        load8f(p1 + kh + 32, sc + 16);
        load8f(p1 + kh + 48, sc + 24);
        load8f(p2 + kh, sf);
        load8f(p2 + kh + 16, sf + 8);
        load8f(p2 + kh + 32, sf + 16);
        load8f(p2 + kh + 48, sf + 24);
#pragma unroll
        for (int i = 0; i < 32; ++i) ar[i] = leakyf(ar[i] * sc[i] + sf[i]);
    }
    v16h a0, a1;
#pragma unroll
    for (int i = 0; i < 16; ++i) {
        a0[i] = (_Float16)ar[i];
        a1[i] = (_Float16)ar[16 + i];
    }

    int nn = lane & 15;
    int kb = hi * 8;
#pragma unroll
    for (int jt = 0; jt < 4; ++jt) {
        int jn = jt * 16;
        const _Float16* wp = &WT[(jn + nn) * WS + kb];
        V16 b0, b1;  // element i -> K = kb + i (i<8), kb + 16 + (i-8) (i>=8) (+32 for b1): table order
        b0.h[0] = ((const v8h*)wp)[0];
        b0.h[1] = ((const v8h*)(wp + 16))[0];
        b1.h[0] = ((const v8h*)(wp + 32))[0];
        b1.h[1] = ((const v8h*)(wp + 48))[0];
        v8f c = {};
        c = __builtin_amdgcn_wmma_f32_16x16x32_f16(false, a0, false, b0.v, (short)0, c, false, false);
        c = __builtin_amdgcn_wmma_f32_16x16x32_f16(false, a1, false, b1.v, (short)0, c, false, false);
        float bv = bias[jn + nn];
#pragma unroll
        for (int r = 0; r < 8; ++r) {  // D: VGPR r -> M = r + 8*hi, N = lane%16
            int orow = tile * 16 + r + hi * 8;
            if (orow < N) {
                float v = c[r] + bv;
                if constexpr (MODE == 1) v = leakyf(v);
                out[(size_t)orow * 64 + jn + nn] = v;
            }
        }
    }
}

// ------------------------------------------------- BN statistics (sum, sumsq per channel)
__global__ void k_stats(const float* __restrict__ h, float* __restrict__ stats, int N) {
    __shared__ float sh[256], sh2[256];
    int c = threadIdx.x & 63;
    int sub = threadIdx.x >> 6;  // 0..3
    float s = 0.f, ss = 0.f;
    for (int n = blockIdx.x * 4 + sub; n < N; n += gridDim.x * 4) {
        float v = h[(size_t)n * 64 + c];
        s += v;
        ss += v * v;
    }
    sh[threadIdx.x] = s;
    sh2[threadIdx.x] = ss;
    __syncthreads();
    if (sub == 0) {
        s = sh[c] + sh[64 + c] + sh[128 + c] + sh[192 + c];
        ss = sh2[c] + sh2[64 + c] + sh2[128 + c] + sh2[192 + c];
        atomicAdd(&stats[c], s);
        atomicAdd(&stats[64 + c], ss);
    }
}

__global__ void k_finalize(const float* __restrict__ stats, const float* __restrict__ g,
                           const float* __restrict__ be, float* __restrict__ scale,
                           float* __restrict__ shift, int N) {
    int c = threadIdx.x;
    if (c >= 64) return;
    float invN = 1.f / (float)N;
    float mu = stats[c] * invN;
    float var = stats[64 + c] * invN - mu * mu;
    float rs = rsqrtf(var + BN_EPS);
    float sc = rs * g[c];
    scale[c] = sc;
    shift[c] = be[c] - mu * sc;
}

// ------------------------------------------------- head prep: transpose Wh -> f16 [512][64],
// zero-pad cols 500..511, pad bh/We with zeros (kills divergence in the hot loop)
__global__ void k_prep_head(const float* __restrict__ Wh, const float* __restrict__ bh,
                            const float* __restrict__ We, _Float16* __restrict__ WhT,
                            float* __restrict__ bhp, float* __restrict__ wep) {
    int t = blockIdx.x * blockDim.x + threadIdx.x;
    if (t >= 512 * 64) return;
    int col = t >> 6, k = t & 63;
    float v = (col < 500) ? Wh[(size_t)k * 500 + col] : 0.f;
    WhT[(size_t)col * 64 + k] = (_Float16)v;
    if (k == 0) {
        bhp[col] = (col < 500) ? bh[col] : 0.f;
        wep[col] = (col < 500) ? We[col] : 0.f;
    }
}

// ------------------------------------------------- head: leaky(X@Wh+bh) @ We + bee, fused
__global__ void k_head(const float* __restrict__ X, const _Float16* __restrict__ WhT,
                       const float* __restrict__ bhp, const float* __restrict__ wep,
                       const float* __restrict__ bee, float* __restrict__ out, int N) {
    int ntiles = (N + 15) >> 4;
    int tile = blockIdx.x * (blockDim.x >> 5) + (threadIdx.x >> 5);
    if (tile >= ntiles) return;
    int lane = threadIdx.x & 31;
    int m = lane & 15;
    int hi = lane >> 4;
    int row = tile * 16 + m;
    if (row >= N) row = N - 1;
    int kh = hi * 8;

    float ar[32];
    const float* r0 = X + (size_t)row * 64;
    load8f(r0 + kh, ar);
    load8f(r0 + kh + 16, ar + 8);
    load8f(r0 + kh + 32, ar + 16);
    load8f(r0 + kh + 48, ar + 24);
    v16h a0, a1;
#pragma unroll
    for (int i = 0; i < 16; ++i) {
        a0[i] = (_Float16)ar[i];
        a1[i] = (_Float16)ar[16 + i];
    }

    int nn = lane & 15;
    int kb = hi * 8;
    float acc[8];
#pragma unroll
    for (int r = 0; r < 8; ++r) acc[r] = 0.f;
    float beev = bee[0];

    for (int jt = 0; jt < 32; ++jt) {  // 512 padded cols (500 real)
        int col = jt * 16 + nn;
        const _Float16* wp = WhT + (size_t)col * 64 + kb;
        V16 b0, b1;  // contiguous halves -> 2x global_load_b128 each
        b0.h[0] = ((const v8h*)wp)[0];
        b0.h[1] = ((const v8h*)(wp + 16))[0];
        b1.h[0] = ((const v8h*)(wp + 32))[0];
        b1.h[1] = ((const v8h*)(wp + 48))[0];
        float wv = wep[col];  // 0 for padded cols: masks them with no branch
        float bv = bhp[col];
        v8f c = {};
        c = __builtin_amdgcn_wmma_f32_16x16x32_f16(false, a0, false, b0.v, (short)0, c, false, false);
        c = __builtin_amdgcn_wmma_f32_16x16x32_f16(false, a1, false, b1.v, (short)0, c, false, false);
#pragma unroll
        for (int r = 0; r < 8; ++r) acc[r] += leakyf(c[r] + bv) * wv;
    }
    // row-sum across the 16-lane group holding each D row
#pragma unroll
    for (int off = 1; off < 16; off <<= 1) {
#pragma unroll
        for (int r = 0; r < 8; ++r) acc[r] += __shfl_xor(acc[r], off, 32);
    }
    if ((lane & 15) == 0) {
#pragma unroll
        for (int r = 0; r < 8; ++r) {
            int orow = tile * 16 + hi * 8 + r;  // lane0 -> rows 0..7, lane16 -> 8..15
            if (orow < N) out[orow] = acc[r] + beev;
        }
    }
}

// ---------------------------------------------------------------- driver
extern "C" void kernel_launch(void* const* d_in, const int* in_sizes, int n_in,
                              void* d_out, int out_size, void* d_ws, size_t ws_size,
                              hipStream_t stream) {
    const float*     x   = (const float*)d_in[0];
    const long long* ei  = (const long long*)d_in[1];  // int64 edge_index [2,E]
    const float*     ea  = (const float*)d_in[2];
    const float*     eW1 = (const float*)d_in[3];
    const float*     eb1 = (const float*)d_in[4];
    const float*     W1a = (const float*)d_in[5];
    const float*     b1a = (const float*)d_in[6];
    const float*     g1  = (const float*)d_in[7];
    const float*     be1 = (const float*)d_in[8];
    const float*     W1b = (const float*)d_in[9];
    const float*     b1b = (const float*)d_in[10];
    const float*     eWs = (const float*)d_in[11];
    const float*     ebs = (const float*)d_in[12];
    const float*     Was = (const float*)d_in[13];
    const float*     bas = (const float*)d_in[14];
    const float*     gs  = (const float*)d_in[15];
    const float*     bes = (const float*)d_in[16];
    const float*     Wbs = (const float*)d_in[17];
    const float*     bbs = (const float*)d_in[18];
    const float*     Wh  = (const float*)d_in[19];
    const float*     bh  = (const float*)d_in[20];
    const float*     We  = (const float*)d_in[21];
    const float*     bee = (const float*)d_in[22];
    float* out = (float*)d_out;

    const int N = in_sizes[0] / 2;  // F_IN = 2
    const int E = in_sizes[2] / 7;  // F_EDGE = 7

    // workspace carving in floats, each chunk 64B aligned
    auto aup = [](size_t v) { return (v + 15) & ~(size_t)15; };
    float* ws = (float*)d_ws;
    size_t off = 0;
    float* agg   = ws + off; off += aup((size_t)N * 64);
    float* hbuf  = ws + off; off += aup((size_t)N * 64);
    float* X     = ws + off; off += aup((size_t)N * 64);
    float* agg2  = ws + off; off += aup((size_t)N * 2);
    float* stats = ws + off; off += 128;
    float* scale = ws + off; off += 64;
    float* shift = ws + off; off += 64;
    float* bhp   = ws + off; off += 512;
    float* wep   = ws + off; off += 512;
    _Float16* WhT = (_Float16*)(ws + off);  // 512*64 halves = 64KB

    const dim3 blk(256);
    const int  ntiles = (N + 15) >> 4;
    const int  gTile  = (ntiles + 7) / 8;  // 8 waves/block, 1 wave per 16-row tile

    // head weight prep (independent of everything else)
    k_prep_head<<<(512 * 64 + 255) / 256, blk, 0, stream>>>(Wh, bh, We, WhT, bhp, wep);

    // ---- layer 1 (in_channels = 2)
    k_zero<<<256, blk, 0, stream>>>(agg2, N * 2);
    k_zero<<<1, 128, 0, stream>>>(stats, 128);
    k_edge2<<<(E + 255) / 256, blk, 0, stream>>>(ei, ea, x, eW1, eb1, agg2, E);
    k_gemm1<<<(N * 64 + 255) / 256, blk, 0, stream>>>(x, agg2, W1a, b1a, hbuf, N);
    k_stats<<<256, blk, 0, stream>>>(hbuf, stats, N);
    k_finalize<<<1, 64, 0, stream>>>(stats, g1, be1, scale, shift, N);
    k_gemm64<1><<<gTile, blk, 0, stream>>>(hbuf, scale, shift, W1b, b1b, X, N);

    // ---- layers 2..5 (in_channels = 64)
    for (int l = 0; l < 4; ++l) {
        k_zero<<<256, blk, 0, stream>>>(agg, N * 64);
        k_zero<<<1, 128, 0, stream>>>(stats, 128);
        k_edge64<<<8192, blk, 0, stream>>>(ei, ea, X, eWs + (size_t)l * 7 * 64,
                                           ebs + l * 64, agg, E);
        k_gemm64<0><<<gTile, blk, 0, stream>>>(X, agg, nullptr, Was + (size_t)l * 64 * 64,
                                               bas + l * 64, hbuf, N);
        k_stats<<<256, blk, 0, stream>>>(hbuf, stats, N);
        k_finalize<<<1, 64, 0, stream>>>(stats, gs + l * 64, bes + l * 64, scale, shift, N);
        k_gemm64<1><<<gTile, blk, 0, stream>>>(hbuf, scale, shift, Wbs + (size_t)l * 64 * 64,
                                               bbs + l * 64, X, N);
    }

    // ---- regression head (both GEMMs fused, no [N,500] intermediate)
    k_head<<<gTile, blk, 0, stream>>>(X, WhT, bhp, wep, bee, out, N);
}
